// SlidingWindowCausalAttention_3066606650062
// MI455X (gfx1250) — matrix operand order repaired and checked by reading, not compiled, not run on hardware
//
#include <hip/hip_runtime.h>

// ---------------------------------------------------------------------------
// Sliding-window causal attention forward for gfx1250 (MI455X).
// B=1, H=16, S=8192, D=64, WINDOW=1024, fp32 in/out.
// Core: v_wmma_f32_16x16x32_bf16, bf16x3 split precision (near-fp32).
// K/V staged into double-buffered LDS pre-converted to packed bf16 hi/lo
// (convert once per chunk, shared by 4 waves); global loads software-
// pipelined two chunks ahead; one workgroup barrier per chunk.
// ---------------------------------------------------------------------------

typedef __attribute__((ext_vector_type(16))) __bf16 v16bf;
typedef __attribute__((ext_vector_type(8)))  __bf16 v8bf;
typedef __attribute__((ext_vector_type(4)))  __bf16 v4bf;
typedef __attribute__((ext_vector_type(8)))  float  v8f;

constexpr int Hh = 16;
constexpr int Ss = 8192;
constexpr int Dd = 64;
constexpr int Ww = 1024;

constexpr int CK    = 32;          // keys per LDS chunk
constexpr int MQ    = 16;          // queries per wave
constexpr int WAVES = 4;
constexpr int QB    = MQ * WAVES;  // 64 queries per workgroup
constexpr int KBROW = Dd + 8;      // bf16 units per K row (144B, 16B-aligned)
constexpr int VBROW = CK + 8;      // bf16 units per V^T row (80B, 16B-aligned)
constexpr int KBUF  = CK * KBROW;  // one K buffer (bf16 units)
constexpr int VBUF  = Dd * VBROW;  // one V^T buffer (bf16 units)

// 1/sqrt(64) * log2(e): scores computed directly in exp2 domain
constexpr float QSCALE = 0.125f * 1.44269504088896340736f;

__device__ __forceinline__ __bf16 f2bf(float x) { return (__bf16)x; }

__device__ __forceinline__ void split4(float a, float b, float c, float d,
                                       v4bf& hi, v4bf& lo) {
    __bf16 h0 = f2bf(a); hi[0] = h0; lo[0] = f2bf(a - (float)h0);
    __bf16 h1 = f2bf(b); hi[1] = h1; lo[1] = f2bf(b - (float)h1);
    __bf16 h2 = f2bf(c); hi[2] = h2; lo[2] = f2bf(c - (float)h2);
    __bf16 h3 = f2bf(d); hi[3] = h3; lo[3] = f2bf(d - (float)h3);
}

#define CAT16(a, b) __builtin_shufflevector(a, b, 0,1,2,3,4,5,6,7,8,9,10,11,12,13,14,15)

__global__ __launch_bounds__(WAVES * 32)
void swa_fwd_kernel(const float* __restrict__ Qp, const float* __restrict__ Kp,
                    const float* __restrict__ Vp, float* __restrict__ Op)
{
    // Double-buffered: K row-major [key][d], V transposed [d][key], bf16 hi/lo.
    __shared__ __bf16 sKhi[2 * KBUF];
    __shared__ __bf16 sKlo[2 * KBUF];
    __shared__ __bf16 sVhi[2 * VBUF];
    __shared__ __bf16 sVlo[2 * VBUF];

    const int tid    = threadIdx.x;
    const int wave   = tid >> 5;
    const int lane   = tid & 31;
    const int halfid = lane >> 4;
    const int n      = lane & 15;

    const int nqblk = Ss / QB;
    const int h     = blockIdx.x / nqblk;
    const int qblk  = blockIdx.x % nqblk;
    const int qbase = qblk * QB;
    const int qt0   = qbase + wave * MQ;  // wave's first query
    const int q     = qt0 + n;            // this lane's query (S^T column)

    const float NEG_INF = -__builtin_inff();

    const float* Kh = Kp + (size_t)h * Ss * Dd;
    const float* Vh = Vp + (size_t)h * Ss * Dd;

    // ---- Q operand (B-layout of S^T = K x Q^T), pre-scaled, bf16 hi/lo ----
    v16bf qhi[2], qlo[2];
    {
        const float* qrow = Qp + ((size_t)h * Ss + q) * Dd;
        #pragma unroll
        for (int c = 0; c < 2; ++c) {
            const float* p = qrow + c * 32 + halfid * 8;
            #pragma unroll
            for (int i = 0; i < 16; ++i) {
                float x   = p[i < 8 ? i : i + 8] * QSCALE;
                __bf16 hb = f2bf(x);
                qhi[c][i] = hb;
                qlo[c][i] = f2bf(x - (float)hb);
            }
        }
    }

    v8f   acc[4] = {};
    float m = NEG_INF;
    float l = 0.0f;

    const int k0   = (qbase >= Ww) ? (qbase - Ww) : 0;
    const int kend = qbase + QB;

    // ---- global fetch of one 32-key chunk into registers ------------------
    float4 kreg[4];
    float  vreg[4][4];
    auto prefetch = [&](int kc) {
        #pragma unroll
        for (int i = 0; i < 4; ++i) {               // K: row-major float4
            int flat = tid + i * 128;
            int row  = flat >> 4;
            int c4   = (flat & 15) << 2;
            kreg[i] = *(const float4*)(Kh + (size_t)(kc + row) * Dd + c4);
        }
        #pragma unroll
        for (int i = 0; i < 4; ++i) {               // V: column-wise (d, 4 keys)
            int flat = tid + i * 128;
            int d    = flat & 63;
            int kg   = flat >> 6;
            #pragma unroll
            for (int j = 0; j < 4; ++j)
                vreg[i][j] = Vh[(size_t)(kc + kg * 4 + j) * Dd + d];
        }
    };

    // ---- convert registers -> packed bf16 hi/lo in LDS buffer `buf` -------
    auto stage = [&](int buf) {
        __bf16* dKhi = sKhi + buf * KBUF;
        __bf16* dKlo = sKlo + buf * KBUF;
        __bf16* dVhi = sVhi + buf * VBUF;
        __bf16* dVlo = sVlo + buf * VBUF;
        #pragma unroll
        for (int i = 0; i < 4; ++i) {
            int flat = tid + i * 128;
            int row  = flat >> 4;
            int c4   = (flat & 15) << 2;
            v4bf hi4, lo4;
            split4(kreg[i].x, kreg[i].y, kreg[i].z, kreg[i].w, hi4, lo4);
            *(v4bf*)(dKhi + row * KBROW + c4) = hi4;
            *(v4bf*)(dKlo + row * KBROW + c4) = lo4;
        }
        #pragma unroll
        for (int i = 0; i < 4; ++i) {
            int flat = tid + i * 128;
            int d    = flat & 63;
            int kg   = flat >> 6;
            v4bf hi4, lo4;
            split4(vreg[i][0], vreg[i][1], vreg[i][2], vreg[i][3], hi4, lo4);
            *(v4bf*)(dVhi + d * VBROW + kg * 4) = hi4;
            *(v4bf*)(dVlo + d * VBROW + kg * 4) = lo4;
        }
    };

    // Pipeline prologue: chunk0 -> LDS buf0; chunk1 -> registers.
    prefetch(k0);
    stage(0);
    if (k0 + CK < kend) prefetch(k0 + CK);
    __syncthreads();

    int ib = 0;
    for (int kc = k0; kc < kend; kc += CK, ib ^= 1) {
        const __bf16* bKhi = sKhi + ib * KBUF;
        const __bf16* bKlo = sKlo + ib * KBUF;
        const __bf16* bVhi = sVhi + ib * VBUF;
        const __bf16* bVlo = sVlo + ib * VBUF;

        // ---- S^T tiles: two 16-key x 16-query tiles, bf16x3 over d=64 ----
        float p0[8], p1[8];
        const bool fullchunk = (kc + CK - 1 <= qt0) && (kc >= qt0 + MQ - 1 - Ww);
        #pragma unroll
        for (int t = 0; t < 2; ++t) {
            v8f s = {};
            #pragma unroll
            for (int c = 0; c < 2; ++c) {
                const __bf16* kb = bKhi + (t * 16 + n) * KBROW + c * 32 + halfid * 8;
                const __bf16* kl = bKlo + (t * 16 + n) * KBROW + c * 32 + halfid * 8;
                v16bf ahi = CAT16(*(const v8bf*)kb, *(const v8bf*)(kb + 16));
                v16bf alo = CAT16(*(const v8bf*)kl, *(const v8bf*)(kl + 16));
                s = __builtin_amdgcn_wmma_f32_16x16x32_bf16(false, ahi, false, qhi[c], (short)0, s, false, false);
                s = __builtin_amdgcn_wmma_f32_16x16x32_bf16(false, ahi, false, qlo[c], (short)0, s, false, false);
                s = __builtin_amdgcn_wmma_f32_16x16x32_bf16(false, alo, false, qhi[c], (short)0, s, false, false);
            }
            float* pt = (t == 0) ? p0 : p1;
            if (fullchunk) {
                #pragma unroll
                for (int r = 0; r < 8; ++r) pt[r] = s[r];
            } else {
                #pragma unroll
                for (int r = 0; r < 8; ++r) {
                    int key  = kc + t * 16 + halfid * 8 + r;
                    int diff = q - key;
                    pt[r] = (diff >= 0 && diff <= Ww) ? s[r] : NEG_INF;
                }
            }
        }

        // ---- Online softmax in exp2 domain (lane = query) ----------------
        float rowmax_lane = p0[0];
        #pragma unroll
        for (int r = 1; r < 8; ++r) rowmax_lane = fmaxf(rowmax_lane, p0[r]);
        #pragma unroll
        for (int r = 0; r < 8; ++r) rowmax_lane = fmaxf(rowmax_lane, p1[r]);
        float rowmax = fmaxf(rowmax_lane, __shfl_xor(rowmax_lane, 16, 32));
        float mnew   = fmaxf(m, rowmax);
        float msafe  = (mnew == NEG_INF) ? 0.0f : mnew;
        float scale  = __builtin_amdgcn_exp2f(m - msafe);
        m = mnew;
        float rs = 0.0f;
        #pragma unroll
        for (int r = 0; r < 8; ++r) {
            p0[r] = __builtin_amdgcn_exp2f(p0[r] - msafe);
            p1[r] = __builtin_amdgcn_exp2f(p1[r] - msafe);
            rs += p0[r] + p1[r];
        }
        l = l * scale + rs + __shfl_xor(rs, 16, 32);

        // ---- P^T as B-operand (32 keys x 16 queries) from own lanes (no swap) -------
        v16bf phi, plo;
        #pragma unroll
        for (int j = 0; j < 8; ++j) {
            float o0  = __shfl_xor(p0[j], 16, 32);
            float o1  = __shfl_xor(p1[j], 16, 32);
            float elo = p0[j];                   // element j   <-> key (halfid?8:0)+j
            float ehi = p1[j];                   // element 8+j <-> key (halfid?24:16)+j
            __bf16 hl = f2bf(elo);
            phi[j]     = hl; plo[j]     = f2bf(elo - (float)hl);
            __bf16 hh = f2bf(ehi);
            phi[8 + j] = hh; plo[8 + j] = f2bf(ehi - (float)hh);
        }

        // ---- out^T += V^T x P^T (4 d-tiles, bf16x3) ----------------------
        #pragma unroll
        for (int td = 0; td < 4; ++td) {
            acc[td] *= scale;
            const __bf16* vb = bVhi + (td * 16 + n) * VBROW + halfid * 8;
            const __bf16* vl = bVlo + (td * 16 + n) * VBROW + halfid * 8;
            v16bf vhi = CAT16(*(const v8bf*)vb, *(const v8bf*)(vb + 16));
            v16bf vlo = CAT16(*(const v8bf*)vl, *(const v8bf*)(vl + 16));
            acc[td] = __builtin_amdgcn_wmma_f32_16x16x32_bf16(false, vhi, false, phi, (short)0, acc[td], false, false);
            acc[td] = __builtin_amdgcn_wmma_f32_16x16x32_bf16(false, vhi, false, plo, (short)0, acc[td], false, false);
            acc[td] = __builtin_amdgcn_wmma_f32_16x16x32_bf16(false, vlo, false, phi, (short)0, acc[td], false, false);
        }

        // ---- pipeline: stage chunk+1 into other buffer, fetch chunk+2 ----
        if (kc + CK < kend) {
            stage(ib ^ 1);
            if (kc + 2 * CK < kend) prefetch(kc + 2 * CK);
        }
        __syncthreads();   // one barrier per chunk (double-buffered)
    }

    // ---- Normalize and store ---------------------------------------------
    float linv = 1.0f / l;
    float* orow = Op + ((size_t)h * Ss + q) * Dd + halfid * 8;
    #pragma unroll
    for (int td = 0; td < 4; ++td) {
        float4 lo4 = { acc[td][0] * linv, acc[td][1] * linv,
                       acc[td][2] * linv, acc[td][3] * linv };
        float4 hi4 = { acc[td][4] * linv, acc[td][5] * linv,
                       acc[td][6] * linv, acc[td][7] * linv };
        *(float4*)(orow + td * 16)     = lo4;
        *(float4*)(orow + td * 16 + 4) = hi4;
    }
}

extern "C" void kernel_launch(void* const* d_in, const int* in_sizes, int n_in,
                              void* d_out, int out_size, void* d_ws, size_t ws_size,
                              hipStream_t stream) {
    const float* Q = (const float*)d_in[0];
    const float* K = (const float*)d_in[1];
    const float* V = (const float*)d_in[2];
    float* O = (float*)d_out;
    dim3 grid(Hh * (Ss / QB));   // 16 heads x 128 query blocks = 2048 WGs
    dim3 block(WAVES * 32);      // 4 waves
    swa_fwd_kernel<<<grid, block, 0, stream>>>(Q, K, V, O);
}
